// SoftHoughTransform_71906342470004
// MI455X (gfx1250) — matrix operand order repaired and checked by reading, not compiled, not run on hardware
//
#include <hip/hip_runtime.h>

// Soft Hough transform as a banded GEMM on the CDNA5 WMMA pipe.
//
//   hough[b,a,r] = sum_p mask[b,p] * max(0, 1 - |rho(a,p) - (r - 363)|)
//
// M = 16 batches, N = 16-bin rho window, K = 32 consecutive x of one image
// row.  B (tent weights) is built with packed-f16 VALU ops that co-execute
// with v_wmma_f32_16x16x32_f16.  One wave owns one (angle, window) tile.
// A pre-pass converts the f32 mask to f16 in d_ws so the hot loop does two
// 16-byte loads per lane and no cvt work.

#define NUM_BATCH 16
#define NUM_ANG   90
#define IMG       256
#define NRHO      727       // 2*363 + 1
#define NWIN      46        // ceil(727 / 16)
#define WPB       8         // waves per 256-thread block
#define MASK_ELEMS (NUM_BATCH * IMG * IMG)

typedef __attribute__((ext_vector_type(16))) _Float16 v16h;
typedef __attribute__((ext_vector_type(8)))  _Float16 v8h;
typedef __attribute__((ext_vector_type(4)))  _Float16 v4h;
typedef __attribute__((ext_vector_type(2)))  _Float16 h2;
typedef __attribute__((ext_vector_type(8)))  float    v8f;

static __device__ __forceinline__ h2 habs2(h2 v) {
  unsigned u = __builtin_bit_cast(unsigned, v) & 0x7FFF7FFFu;  // v_and_b32
  return __builtin_bit_cast(h2, u);
}

// ---- pre-pass: f32 mask -> f16 mask (same [b][y][x] layout) --------------
__global__ __launch_bounds__(256)
void cvt_mask_f16_kernel(const float* __restrict__ in,
                         _Float16* __restrict__ out) {
  const int i = blockIdx.x * blockDim.x + threadIdx.x;   // one float4 each
  if (i < MASK_ELEMS / 4) {
    const float4 v = ((const float4*)in)[i];
    v4h o;
    o[0] = (_Float16)v.x; o[1] = (_Float16)v.y;
    o[2] = (_Float16)v.z; o[3] = (_Float16)v.w;
    ((v4h*)out)[i] = o;
  }
}

// ---- main banded-GEMM kernel --------------------------------------------
template <bool F16SRC>
__global__ __launch_bounds__(256)
void SoftHough_wmma_kernel(const float* __restrict__ mask,
                           const _Float16* __restrict__ maskh,
                           float* __restrict__ out) {
  const int lane = threadIdx.x & 31;
  const int wave = threadIdx.x >> 5;
  const int tile = blockIdx.x * WPB + wave;
  if (tile >= NUM_ANG * NWIN) return;          // wave-uniform guard

  const int ang = tile / NWIN;
  const int win = tile - ang * NWIN;
  const int r0  = win * 16;

  const float theta = 3.14159265358979323846f * (float)ang / (float)NUM_ANG;
  float sn, cs;
  __sincosf(theta, &sn, &cs);

  // --- lane roles per CDNA5 16-bit WMMA VGPR layouts ---------------------
  const int m    = lane & 15;              // A: batch (M); B: bin col (N); D: N
  const int kb_a = (lane < 16) ? 0 : 16;   // A K-offset
  const int kb_b = (lane < 16) ? 0 : 16;   // B K-offset
  const float nf = (float)m;

  const float*    mimg  = mask  + (size_t)m * (IMG * IMG);
  const _Float16* mimgh = maskh + (size_t)m * (IMG * IMG);

  const h2 step2 = {(_Float16)(2.f * cs), (_Float16)(2.f * cs)};
  const h2 ones  = {(_Float16)1.f, (_Float16)1.f};
  const h2 zeros = {(_Float16)0.f, (_Float16)0.f};

  v8f acc = {};                            // D[16x16] f32 accumulator

  for (int y = 0; y < IMG; ++y) {
    // rho(x) = rowbase + x*cs   (bin units; normalization scale is exactly 1)
    const float rowbase = ((float)y - 128.f) * sn + 363.f - 128.f * cs;

    // x-range with rho in (r0-1, r0+16)  -- uniform across the wave
    int cx0, cx1;
    const float ac = fabsf(cs);
    if (ac > 1e-4f) {
      const float x1 = ((float)(r0 - 1)  - rowbase) / cs;
      const float x2 = ((float)(r0 + 16) - rowbase) / cs;
      const float xlo = fminf(x1, x2);
      const float xhi = fmaxf(x1, x2);
      if (xhi < 0.f || xlo > 255.f) continue;
      int ilo = (int)xlo;
      int ihi = (int)fminf(fmaxf(xhi, 0.f), 255.f);
      cx0 = (ilo < 0) ? 0 : (ilo >> 5);
      cx1 = ihi >> 5;
      if (cx1 > 7) cx1 = 7;
    } else {                               // theta ~ 90 deg: row -> ~1 bin
      if (rowbase <= (float)(r0 - 1) || rowbase >= (float)(r0 + 16)) continue;
      cx0 = 0; cx1 = 7;
    }

    const float*    row  = mimg  + y * IMG;
    const _Float16* rowh = mimgh + y * IMG;

    for (int cx = cx0; cx <= cx1; ++cx) {
      const int x0 = cx << 5;              // aligned 32-pixel chunk

      // ---- A tile: mask[batch=m, y, x0..x0+31], contiguous k = 16h + s -----
      v16h Ah;
      if constexpr (F16SRC) {
        const v8h a0 = *(const v8h*)(rowh + x0 + kb_a);        // K = kb_a..+7
        const v8h a1 = *(const v8h*)(rowh + x0 + 8 + kb_a);    // K = 8+kb_a..+7
#pragma unroll
        for (int i = 0; i < 8; ++i) { Ah[i] = a0[i]; Ah[8 + i] = a1[i]; }
      } else {
        const float4* p0 = (const float4*)(row + x0 + kb_a);
        const float4* p1 = (const float4*)(row + x0 + 8 + kb_a);
        const float4 a0 = p0[0], a1 = p0[1];
        const float4 b0 = p1[0], b1 = p1[1];
        Ah[0]  = (_Float16)a0.x; Ah[1]  = (_Float16)a0.y;
        Ah[2]  = (_Float16)a0.z; Ah[3]  = (_Float16)a0.w;
        Ah[4]  = (_Float16)a1.x; Ah[5]  = (_Float16)a1.y;
        Ah[6]  = (_Float16)a1.z; Ah[7]  = (_Float16)a1.w;
        Ah[8]  = (_Float16)b0.x; Ah[9]  = (_Float16)b0.y;
        Ah[10] = (_Float16)b0.z; Ah[11] = (_Float16)b0.w;
        Ah[12] = (_Float16)b1.x; Ah[13] = (_Float16)b1.y;
        Ah[14] = (_Float16)b1.z; Ah[15] = (_Float16)b1.w;
      }

      // ---- B tile: tent weights, packed-f16 incremental build ----------
      //   t_e = tbase + e*cs ;  w_e = max(0, 1 - |t_e|)
      const float tbase =
          fmaf((float)(x0 + kb_b), cs, rowbase) - (float)r0 - nf;
      h2 tcur = {(_Float16)tbase, (_Float16)(tbase + cs)};
      v16h Bh;
#pragma unroll
      for (int j = 0; j < 8; ++j) {
        const h2 a = habs2(tcur);                              // v_and_b32
        const h2 w = __builtin_elementwise_max(ones - a, zeros);  // pk_add/pk_max
        Bh[2 * j]     = w[0];
        Bh[2 * j + 1] = w[1];
        tcur = tcur + step2;                                   // v_pk_add_f16
      }

      // D += A x B   (v_wmma_f32_16x16x32_f16)
      acc = __builtin_amdgcn_wmma_f32_16x16x32_f16(
          /*neg_a=*/false, Ah, /*neg_b=*/false, Bh,
          /*c_mod=*/(short)0, acc, /*reuse_a=*/false, /*reuse_b=*/false);

      // prefetch same chunk of the next image row (global_prefetch_b8)
      if constexpr (F16SRC)
        __builtin_prefetch((const void*)(rowh + IMG + x0), 0, 0);
      else
        __builtin_prefetch((const void*)(row + IMG + x0), 0, 0);
    }
  }

  // ---- store D: lane&15 = N (bin), VGPR v -> M = v + 8*(lane>=16) -------
  const int mb  = (lane >> 4) * 8;
  const int bin = r0 + m;
  if (bin < NRHO) {
    float* obase = out + (size_t)ang * NRHO + bin;
#pragma unroll
    for (int v = 0; v < 8; ++v) {
      obase[(size_t)(mb + v) * (NUM_ANG * NRHO)] = acc[v];
    }
  }
}

extern "C" void kernel_launch(void* const* d_in, const int* in_sizes, int n_in,
                              void* d_out, int out_size, void* d_ws, size_t ws_size,
                              hipStream_t stream) {
  (void)in_sizes; (void)n_in; (void)out_size;
  const float* mask = (const float*)d_in[0];   // [16,1,256,256] f32
  float* out = (float*)d_out;                  // [16,90,727] f32

  const int tiles  = NUM_ANG * NWIN;           // 4140 wave-tiles
  const int blocks = (tiles + WPB - 1) / WPB;  // 518 blocks x 256 threads

  const size_t f16_bytes = (size_t)MASK_ELEMS * sizeof(_Float16);  // 2 MB
  if (ws_size >= f16_bytes) {
    _Float16* maskh = (_Float16*)d_ws;
    cvt_mask_f16_kernel<<<MASK_ELEMS / 4 / 256, 256, 0, stream>>>(mask, maskh);
    SoftHough_wmma_kernel<true><<<blocks, 256, 0, stream>>>(mask, maskh, out);
  } else {
    SoftHough_wmma_kernel<false><<<blocks, 256, 0, stream>>>(mask, nullptr, out);
  }
}
